// PromptSelfAttention_19559281066086
// MI455X (gfx1250) — matrix operand order repaired and checked by reading, not compiled, not run on hardware
//
#include <hip/hip_runtime.h>

// ---------------------------------------------------------------------------
// Types
// ---------------------------------------------------------------------------
typedef __attribute__((ext_vector_type(16))) _Float16 v16h;
typedef __attribute__((ext_vector_type(8)))  _Float16 v8h;
typedef __attribute__((ext_vector_type(4)))  _Float16 v4h;
typedef __attribute__((ext_vector_type(8)))  float    v8f;
typedef __attribute__((ext_vector_type(4)))  float    v4f;
typedef __attribute__((ext_vector_type(4)))  int      v4i;

#define WMMA_F32_F16(a, b, c) \
  __builtin_amdgcn_wmma_f32_16x16x32_f16(false, (a), false, (b), (short)0, (c), false, false)

// ---------------------------------------------------------------------------
// Async global->LDS staging (CDNA5 GLOBAL_LOAD_ASYNC_TO_LDS_B128, ASYNCcnt).
// Signature (verified via round-2 diagnostic):
//   (int4 addrspace(1)* gsrc, int4 addrspace(3)* ldst, int offset, int cpol)
// Falls back to synchronous b128 copies if the builtin is absent.
// ---------------------------------------------------------------------------
#if defined(__has_builtin)
#if __has_builtin(__builtin_amdgcn_global_load_async_to_lds_b128)
#define ASYNC_LDS 1
#endif
#endif

#ifdef ASYNC_LDS
typedef __attribute__((address_space(1))) v4i gas_v4i;
typedef __attribute__((address_space(3))) v4i las_v4i;
#endif

__device__ __forceinline__ void stage16(const _Float16* g, _Float16* l) {
#ifdef ASYNC_LDS
  __builtin_amdgcn_global_load_async_to_lds_b128(
      (gas_v4i*)(void*)(const void*)g, (las_v4i*)(void*)l, 0, 0);
#else
  *(v8h*)l = *(const v8h*)g;
#endif
}

__device__ __forceinline__ void stage_wait() {
#ifdef ASYNC_LDS
#if __has_builtin(__builtin_amdgcn_s_wait_asynccnt)
  __builtin_amdgcn_s_wait_asynccnt(0);
#else
  asm volatile("s_wait_asynccnt 0" ::: "memory");
#endif
#endif
}

// ---------------------------------------------------------------------------
// Fragment loaders (wave32, V_WMMA_F32_16X16X32_F16 layouts per CDNA5 ISA 7.12.2)
// A (16x32): lane holds row m=L%16, h=L/16; halves = two contiguous 8-half
//   runs at K offsets 8h and 16+8h  -> two aligned ds_load_b128.
// B (32x16): lane holds col n=L%16, h=L/16; halves at K offsets 8h and 16+8h
//   (same as A) when the tile is stored column-major (tileT[n][k]) -> two aligned ds_load_b128.
// ---------------------------------------------------------------------------
__device__ __forceinline__ v16h load_a_frag(const _Float16* base, int stride, int kk) {
  int lane = threadIdx.x & 31;
  int m = lane & 15, h = lane >> 4;
  const _Float16* p = base + m * stride + kk + 8 * h;
  v8h lo = *(const v8h*)p;
  v8h hi = *(const v8h*)(p + 16);
  return __builtin_shufflevector(lo, hi, 0, 1, 2, 3, 4, 5, 6, 7,
                                 8, 9, 10, 11, 12, 13, 14, 15);
}

__device__ __forceinline__ v16h load_b_frag(const _Float16* baseT, int stride, int n0, int kk) {
  int lane = threadIdx.x & 31;
  int n = n0 + (lane & 15), h = lane >> 4;
  const _Float16* p = baseT + n * stride + kk + 8 * h;
  v8h lo = *(const v8h*)p;
  v8h hi = *(const v8h*)(p + 16);
  return __builtin_shufflevector(lo, hi, 0, 1, 2, 3, 4, 5, 6, 7,
                                 8, 9, 10, 11, 12, 13, 14, 15);
}

// ---------------------------------------------------------------------------
// 1) LayerNorm (fp32 in) -> f16 activations.  One block per row of 1024.
// ---------------------------------------------------------------------------
__global__ void ln_f16_kernel(const float* __restrict__ x,
                              const float* __restrict__ gamma,
                              const float* __restrict__ beta,
                              _Float16* __restrict__ y) {
  const int DIM = 1024;
  int row = blockIdx.x;
  int t = threadIdx.x;  // 256 threads, 4 elements each
  const float* xr = x + (size_t)row * DIM;
  v4f v = *(const v4f*)(xr + t * 4);
  float s = v[0] + v[1] + v[2] + v[3];
  float s2 = v[0] * v[0] + v[1] * v[1] + v[2] * v[2] + v[3] * v[3];
#pragma unroll
  for (int off = 16; off > 0; off >>= 1) {
    s += __shfl_xor(s, off, 32);
    s2 += __shfl_xor(s2, off, 32);
  }
  __shared__ float redS[8], redS2[8];
  if ((t & 31) == 0) { redS[t >> 5] = s; redS2[t >> 5] = s2; }
  __syncthreads();
  float S = 0.f, S2 = 0.f;
#pragma unroll
  for (int i = 0; i < 8; ++i) { S += redS[i]; S2 += redS2[i]; }
  float mu = S * (1.0f / DIM);
  float var = S2 * (1.0f / DIM) - mu * mu;
  float rstd = rsqrtf(var + 1e-5f);
  v4f g = *(const v4f*)(gamma + t * 4);
  v4f bb = *(const v4f*)(beta + t * 4);
  v4h o;
#pragma unroll
  for (int i = 0; i < 4; ++i)
    o[i] = (_Float16)(((v[i] - mu) * rstd) * g[i] + bb[i]);
  *(v4h*)(y + (size_t)row * DIM + t * 4) = o;
}

// ---------------------------------------------------------------------------
// 2) Weight fp32 [K,N] -> f16 TRANSPOSED [N,K] (done once; makes GEMM B
//    staging contiguous, no LDS scalar transpose).
// ---------------------------------------------------------------------------
__global__ void w_to_f16t_kernel(const float* __restrict__ W,
                                 _Float16* __restrict__ Wt, int K, int N) {
  int n = blockIdx.x * 64 + (threadIdx.x & 63);
  int k = blockIdx.y * 4 + (threadIdx.x >> 6);
  Wt[(size_t)n * K + k] = (_Float16)W[(size_t)k * N + n];
}

// ---------------------------------------------------------------------------
// 3) f16 GEMM: C = alpha * A[M,K] x B[K,N], with B given pre-transposed as
//    Bt[N,K].  Block tile 128x128 (grid = M/128 x N/128), 256 threads =
//    8 waves arranged 4x2, wave tile 32x64 (2x4 accumulators = 8 WMMAs per
//    k-step against 12 fragment ds_load_b128s), LDS K-step 64.
//    OUT_MODE: 0 = f16 row-major, 1 = f32 row-major,
//              2 = f16 scattered to [b, h, d, s] (V for attention),
//                  index = (row>>11)*1024*2048 + col*2048 + (row & 2047)
// ---------------------------------------------------------------------------
template <int OUT_MODE>
__global__ void gemm_f16(const _Float16* __restrict__ A,
                         const _Float16* __restrict__ Bt,
                         void* __restrict__ Cout,
                         int M, int N, int K, float alpha) {
  const int LDT = 72;  // f16 units; 144B rows, 16B aligned
  __shared__ __align__(16) _Float16 As[128 * LDT];
  __shared__ __align__(16) _Float16 Bs[128 * LDT];  // [n][k]
  int t = threadIdx.x;          // 256 threads
  int wave = t >> 5, lane = t & 31;
  int bm = blockIdx.x * 128;
  int bn = blockIdx.y * 128;
  int wm = (wave >> 1) * 32;    // 4 wave-rows
  int wn = (wave & 1) * 64;     // 2 wave-cols

  v8f acc[2][4] = {};

  int r = t >> 1, c = (t & 1) * 32;  // staging coords: 32 halves/thread/tile
  for (int kk = 0; kk < K; kk += 64) {
    __syncthreads();
    {
      const _Float16* ga = A + (size_t)(bm + r) * K + kk + c;
      stage16(ga + 0,  &As[r * LDT + c + 0]);
      stage16(ga + 8,  &As[r * LDT + c + 8]);
      stage16(ga + 16, &As[r * LDT + c + 16]);
      stage16(ga + 24, &As[r * LDT + c + 24]);
      const _Float16* gb = Bt + (size_t)(bn + r) * K + kk + c;
      stage16(gb + 0,  &Bs[r * LDT + c + 0]);
      stage16(gb + 8,  &Bs[r * LDT + c + 8]);
      stage16(gb + 16, &Bs[r * LDT + c + 16]);
      stage16(gb + 24, &Bs[r * LDT + c + 24]);
    }
    stage_wait();
    __syncthreads();

#pragma unroll
    for (int ks = 0; ks < 64; ks += 32) {
      v16h a0 = load_a_frag(&As[(wm + 0) * LDT], LDT, ks);
      v16h a1 = load_a_frag(&As[(wm + 16) * LDT], LDT, ks);
#pragma unroll
      for (int ni = 0; ni < 4; ++ni) {
        v16h b = load_b_frag(Bs, LDT, wn + ni * 16, ks);
        acc[0][ni] = WMMA_F32_F16(a0, b, acc[0][ni]);
        acc[1][ni] = WMMA_F32_F16(a1, b, acc[1][ni]);
      }
    }
  }

  // Epilogue.  C/D layout: element r of lane L -> row (r + 8*(L/16)), col L%16.
  int hl = lane >> 4, n = lane & 15;
#pragma unroll
  for (int mi = 0; mi < 2; ++mi) {
#pragma unroll
    for (int ni = 0; ni < 4; ++ni) {
#pragma unroll
      for (int rr = 0; rr < 8; ++rr) {
        int row = bm + wm + mi * 16 + rr + 8 * hl;
        int col = bn + wn + ni * 16 + n;
        float val = acc[mi][ni][rr] * alpha;
        if (OUT_MODE == 1) {
          ((float*)Cout)[(size_t)row * N + col] = val;
        } else if (OUT_MODE == 2) {
          // V transposed per head: [b, h*64+d, s] with S = 2048
          size_t idx = (((size_t)(row >> 11) * 1024 + col) << 11) + (row & 2047);
          ((_Float16*)Cout)[idx] = (_Float16)val;
        } else {
          ((_Float16*)Cout)[(size_t)row * N + col] = (_Float16)val;
        }
      }
    }
  }
}

// ---------------------------------------------------------------------------
// 4) Flash-style attention.  Q/K: [B, S, H*Dh] f16 (scale folded in);
//    Vt: [B, H*Dh, S] f16 (pre-transposed by projection epilogue).
//    grid = (S/64, B*H), 128 threads = 4 waves; wave owns 16 query rows;
//    loop over 32 K/V tiles of 64 keys.
// ---------------------------------------------------------------------------
__global__ void attn_kernel(const _Float16* __restrict__ Q,
                            const _Float16* __restrict__ K,
                            const _Float16* __restrict__ Vt,
                            _Float16* __restrict__ O) {
  const int S = 2048, ROWSTRIDE = 1024;  // H*Dh
  const int LDT = 72;
  __shared__ __align__(16) _Float16 Qs[64 * LDT];
  __shared__ __align__(16) _Float16 Ks[64 * LDT];
  __shared__ __align__(16) _Float16 VsT[64 * LDT];   // [d][key]
  __shared__ __align__(16) _Float16 Ps[4][16 * LDT]; // per-wave P tile

  int t = threadIdx.x;
  int wave = t >> 5, lane = t & 31;
  int hh = lane >> 4, n16 = lane & 15;
  int bh = blockIdx.y;
  int b = bh >> 4, h = bh & 15;
  int q0 = blockIdx.x * 64;
  const size_t bhoff = (size_t)b * S * ROWSTRIDE + (size_t)h * 64;
  const _Float16* Vbase = Vt + ((size_t)b * 1024 + (size_t)h * 64) * S;

  int r = t >> 1, c = (t & 1) * 32;  // staging coords
  {  // Q tile 64x64 (once)
    const _Float16* g = Q + bhoff + (size_t)(q0 + r) * ROWSTRIDE + c;
    stage16(g + 0,  &Qs[r * LDT + c + 0]);
    stage16(g + 8,  &Qs[r * LDT + c + 8]);
    stage16(g + 16, &Qs[r * LDT + c + 16]);
    stage16(g + 24, &Qs[r * LDT + c + 24]);
  }

  v8f o_acc[4] = {};
  float Mrow[8], Lrow[8];
#pragma unroll
  for (int i = 0; i < 8; ++i) { Mrow[i] = -1e30f; Lrow[i] = 0.f; }

  for (int j = 0; j < S / 64; ++j) {
    __syncthreads();
    {  // K tile, row-major (transposed-B source for Q*K^T)
      const _Float16* g = K + bhoff + (size_t)(j * 64 + r) * ROWSTRIDE + c;
      stage16(g + 0,  &Ks[r * LDT + c + 0]);
      stage16(g + 8,  &Ks[r * LDT + c + 8]);
      stage16(g + 16, &Ks[r * LDT + c + 16]);
      stage16(g + 24, &Ks[r * LDT + c + 24]);
    }
    {  // V tile: Vt rows are d (contiguous in key) -> contiguous staging
      const _Float16* g = Vbase + (size_t)r * S + j * 64 + c;
      stage16(g + 0,  &VsT[r * LDT + c + 0]);
      stage16(g + 8,  &VsT[r * LDT + c + 8]);
      stage16(g + 16, &VsT[r * LDT + c + 16]);
      stage16(g + 24, &VsT[r * LDT + c + 24]);
    }
    if (j + 1 < S / 64) {  // prefetch next tiles (global_prefetch_b8)
      __builtin_prefetch(K + bhoff + (size_t)((j + 1) * 64 + r) * ROWSTRIDE + c, 0, 1);
      __builtin_prefetch(Vbase + (size_t)r * S + (j + 1) * 64 + c, 0, 1);
    }
    stage_wait();
    __syncthreads();

    // S = Q K^T : 16 rows x 64 keys per wave, k-dim = Dh = 64
    v8f s_acc[4] = {};
#pragma unroll
    for (int kk = 0; kk < 64; kk += 32) {
      v16h aq = load_a_frag(&Qs[(wave * 16) * LDT], LDT, kk);
#pragma unroll
      for (int nt = 0; nt < 4; ++nt) {
        v16h bk = load_b_frag(Ks, LDT, nt * 16, kk);
        s_acc[nt] = WMMA_F32_F16(aq, bk, s_acc[nt]);
      }
    }

    // Online softmax (row m = rr + 8*hh; its 16 cols live in one 16-lane half)
#pragma unroll
    for (int rr = 0; rr < 8; ++rr) {
      float mx = fmaxf(fmaxf(s_acc[0][rr], s_acc[1][rr]),
                       fmaxf(s_acc[2][rr], s_acc[3][rr]));
#pragma unroll
      for (int off = 8; off > 0; off >>= 1) mx = fmaxf(mx, __shfl_xor(mx, off, 16));
      float mnew = fmaxf(Mrow[rr], mx);
      float cf = __expf(Mrow[rr] - mnew);
      float p0 = __expf(s_acc[0][rr] - mnew);
      float p1 = __expf(s_acc[1][rr] - mnew);
      float p2 = __expf(s_acc[2][rr] - mnew);
      float p3 = __expf(s_acc[3][rr] - mnew);
      float rs = p0 + p1 + p2 + p3;
#pragma unroll
      for (int off = 8; off > 0; off >>= 1) rs += __shfl_xor(rs, off, 16);
      Lrow[rr] = Lrow[rr] * cf + rs;
      Mrow[rr] = mnew;
      int mloc = rr + 8 * hh;
      Ps[wave][mloc * LDT + 0 * 16 + n16] = (_Float16)p0;
      Ps[wave][mloc * LDT + 1 * 16 + n16] = (_Float16)p1;
      Ps[wave][mloc * LDT + 2 * 16 + n16] = (_Float16)p2;
      Ps[wave][mloc * LDT + 3 * 16 + n16] = (_Float16)p3;
#pragma unroll
      for (int dt = 0; dt < 4; ++dt) o_acc[dt][rr] *= cf;
    }

    // O += P V : Ps is wave-private; per-wave LDS ordering suffices
#pragma unroll
    for (int kk = 0; kk < 64; kk += 32) {
      v16h ap = load_a_frag(&Ps[wave][0], LDT, kk);
#pragma unroll
      for (int dt = 0; dt < 4; ++dt) {
        v16h bv = load_b_frag(VsT, LDT, dt * 16, kk);
        o_acc[dt] = WMMA_F32_F16(ap, bv, o_acc[dt]);
      }
    }
  }

  // Normalize and store f16 attention output (row-major [B,S,H*Dh])
  _Float16* og = O + bhoff + (size_t)(q0 + wave * 16) * ROWSTRIDE;
#pragma unroll
  for (int rr = 0; rr < 8; ++rr) {
    float inv = 1.0f / Lrow[rr];
    int mloc = rr + 8 * hh;
#pragma unroll
    for (int dt = 0; dt < 4; ++dt)
      og[(size_t)mloc * ROWSTRIDE + dt * 16 + n16] = (_Float16)(o_acc[dt][rr] * inv);
  }
}

// ---------------------------------------------------------------------------
// Launcher
// ---------------------------------------------------------------------------
extern "C" void kernel_launch(void* const* d_in, const int* in_sizes, int n_in,
                              void* d_out, int out_size, void* d_ws, size_t ws_size,
                              hipStream_t stream) {
  (void)in_sizes; (void)n_in; (void)out_size;
  const int B = 4, S = 2048, DIM = 1024;
  const size_t MROWS = (size_t)B * S;        // 8192
  const size_t ACT = MROWS * DIM;            // 8M elems
  const size_t WEL = (size_t)DIM * DIM;      // 1M elems

  const float* latents = (const float*)d_in[0];
  const float* gamma   = (const float*)d_in[1];
  const float* beta    = (const float*)d_in[2];
  const float* Wq      = (const float*)d_in[3];
  const float* Wk      = (const float*)d_in[4];
  const float* Wv      = (const float*)d_in[5];
  const float* Wo      = (const float*)d_in[6];
  float* out = (float*)d_out;

  // Workspace (f16): Xln | WqT WkT WvT WoT | Q K Vt | Oattn  (~88 MB)
  char* ws = (char*)d_ws;
  size_t need = (ACT + 4 * WEL + 4 * ACT) * sizeof(_Float16);
  if (ws_size < need) return;  // defensive: avoid OOB scratch writes
  _Float16* Xln = (_Float16*)ws;  ws += ACT * sizeof(_Float16);
  _Float16* Wqt = (_Float16*)ws;  ws += WEL * sizeof(_Float16);
  _Float16* Wkt = (_Float16*)ws;  ws += WEL * sizeof(_Float16);
  _Float16* Wvt = (_Float16*)ws;  ws += WEL * sizeof(_Float16);
  _Float16* Wot = (_Float16*)ws;  ws += WEL * sizeof(_Float16);
  _Float16* Qb  = (_Float16*)ws;  ws += ACT * sizeof(_Float16);
  _Float16* Kb  = (_Float16*)ws;  ws += ACT * sizeof(_Float16);
  _Float16* Vtb = (_Float16*)ws;  ws += ACT * sizeof(_Float16);
  _Float16* Ob  = (_Float16*)ws;

  // 1) LayerNorm -> f16
  ln_f16_kernel<<<(int)MROWS, 256, 0, stream>>>(latents, gamma, beta, Xln);

  // 2) weight convert + transpose (f16 [N,K])
  dim3 wg(DIM / 64, DIM / 4);
  w_to_f16t_kernel<<<wg, 256, 0, stream>>>(Wq, Wqt, DIM, DIM);
  w_to_f16t_kernel<<<wg, 256, 0, stream>>>(Wk, Wkt, DIM, DIM);
  w_to_f16t_kernel<<<wg, 256, 0, stream>>>(Wv, Wvt, DIM, DIM);
  w_to_f16t_kernel<<<wg, 256, 0, stream>>>(Wo, Wot, DIM, DIM);

  // 3) projections (softmax scale 64^-0.25 folded into Q and K);
  //    V written directly in [B, H*Dh, S] transposed layout (OUT_MODE=2)
  const float QK_SCALE = 0.35355339059327373f;
  dim3 gg((unsigned)(MROWS / 128), (unsigned)(DIM / 128));
  gemm_f16<0><<<gg, 256, 0, stream>>>(Xln, Wqt, Qb, (int)MROWS, DIM, DIM, QK_SCALE);
  gemm_f16<0><<<gg, 256, 0, stream>>>(Xln, Wkt, Kb, (int)MROWS, DIM, DIM, QK_SCALE);
  gemm_f16<2><<<gg, 256, 0, stream>>>(Xln, Wvt, Vtb, (int)MROWS, DIM, DIM, 1.0f);

  // 4) attention
  attn_kernel<<<dim3(S / 64, B * 16), 128, 0, stream>>>(Qb, Kb, Vtb, Ob);

  // 5) output projection -> fp32 d_out
  gemm_f16<1><<<gg, 256, 0, stream>>>(Ob, Wot, out, (int)MROWS, DIM, DIM, 1.0f);
}
